// Attention_24438363914913
// MI455X (gfx1250) — matrix operand order repaired and checked by reading, not compiled, not run on hardware
//
#include <hip/hip_runtime.h>
#include <hip/hip_bf16.h>

// ---------------- problem constants (B,T,DIM,H,HD) ----------------
#define BB   2
#define TT   2048
#define DIMM 2048
#define HH   16
#define HDD  128
#define MROWS (BB * TT)        // 4096 token rows
#define NQKV  (3 * HH * HDD)   // 6144

// ---------------- WMMA types / helpers ----------------
typedef __bf16 bf16_t;
typedef bf16_t v16bf __attribute__((ext_vector_type(16)));
typedef float  v8f   __attribute__((ext_vector_type(8)));

__device__ __forceinline__ unsigned short f32_to_bf16_bits(float f) {
    unsigned int u = __builtin_bit_cast(unsigned int, f);
    u += 0x7FFFu + ((u >> 16) & 1u);            // round-to-nearest-even
    return (unsigned short)(u >> 16);
}
__device__ __forceinline__ bf16_t us_to_bf16(unsigned short s) {
    return __builtin_bit_cast(bf16_t, s);
}
__device__ __forceinline__ v8f zero8() {
    v8f r;
    #pragma unroll
    for (int i = 0; i < 8; ++i) r[i] = 0.0f;
    return r;
}
__device__ __forceinline__ v8f wmma_bf16(v16bf a, v16bf b, v8f c) {
    // D = A(16x32 bf16) x B(32x16 bf16) + C(16x16 f32)
    return __builtin_amdgcn_wmma_f32_16x16x32_bf16(
        /*neg_a=*/false, a, /*neg_b=*/false, b,
        /*c_mod=*/(short)0, c, /*reuse_a=*/false, /*reuse_b=*/false);
}
// A-fragment (16x32, 16-bit) K index for slot j (0..15), contiguous order
// matching B: lanes 0-15 hold K0..15, lanes 16-31 K16..31 -> k = hi*16 + j.
__device__ __forceinline__ int a_kidx(int j, int hi) {
    return hi * 16 + j;
}
// B-fragment (32x16, 16-bit), order used here: lanes 0-15 hold K0..15, lanes 16-31 K16..31;
// VGPR j' packs K=2j',2j'+1  ->  slot j maps to k = hi*16 + j.

// Async copy 16B global -> LDS (CDNA5 GLOBAL_LOAD_ASYNC_TO_LDS_B128).
// ldsOff = byte offset within workgroup LDS (low 32 bits of generic ptr).
__device__ __forceinline__ void async_copy16(const void* gaddr, unsigned ldsOff) {
    asm volatile("global_load_async_to_lds_b128 %0, %1, off"
                 :: "v"(ldsOff), "v"(gaddr) : "memory");
}
__device__ __forceinline__ void async_wait0() {
    asm volatile("s_wait_asynccnt 0" ::: "memory");
}
__device__ __forceinline__ unsigned lds_off(const void* p) {
    return (unsigned)(unsigned long long)p;   // LDS generic ptr: low 32b = offset
}

// ---------------- convert f32 -> bf16 ----------------
__global__ void f32_to_bf16_kernel(const float* __restrict__ in,
                                   unsigned short* __restrict__ out, int n) {
    int i = blockIdx.x * blockDim.x + threadIdx.x;
    if (i < n) out[i] = f32_to_bf16_bits(in[i]);
}

// ---------------- tiled bf16 WMMA GEMM ----------------
// C[M,N] = A[M,K] * B[K,N]; A,B bf16(bits); C bf16(bits) or f32.
// Block: 128x64 output, 128 threads (4 waves); wave = 32 rows x 64 cols
// (2 M-frags x 4 N-frags = 8 WMMA per K-step, B frags reused across M).
#define BM 128
#define BN 64
#define BKT 32
#define SA 48   // LDS row stride (ushort) for A tile; 96B rows  -> 16B aligned
#define SB 72   // LDS row stride (ushort) for B tile; 144B rows -> 16B aligned

template <bool OUT_BF16>
__global__ __launch_bounds__(128)
void gemm_wmma_bf16(const unsigned short* __restrict__ A,
                    const unsigned short* __restrict__ Bm,
                    void* __restrict__ Cp,
                    int Mdim, int Ndim, int Kdim) {
    __shared__ unsigned short lA[BM * SA];   // 128x32 tile (padded)
    __shared__ unsigned short lB[BKT * SB];  // 32x64 tile (padded)

    const int tid  = threadIdx.x;
    const int lane = tid & 31;
    const int wave = tid >> 5;
    const int hi   = lane >> 4;
    const int ln   = lane & 15;
    const int rowBase = blockIdx.y * BM;
    const int colBase = blockIdx.x * BN;

    v8f acc[2][4];
    #pragma unroll
    for (int mi = 0; mi < 2; ++mi)
        #pragma unroll
        for (int nc = 0; nc < 4; ++nc) acc[mi][nc] = zero8();

    for (int k0 = 0; k0 < Kdim; k0 += BKT) {
        __syncthreads();   // protect LDS from previous iteration's readers
        // A tile 128x32: 512 16B-chunks, 4 per thread (async direct-to-LDS)
        #pragma unroll
        for (int i = 0; i < 4; ++i) {
            int e = tid + i * 128;
            int r = e >> 2;                 // 4 chunks per 32-col row
            int c = (e & 3) * 8;
            async_copy16(A + (size_t)(rowBase + r) * Kdim + k0 + c,
                         lds_off(&lA[r * SA + c]));
        }
        // B tile 32x64: 256 16B-chunks, 2 per thread
        #pragma unroll
        for (int i = 0; i < 2; ++i) {
            int e = tid + i * 128;
            int r = e >> 3;                 // 8 chunks per 64-col row
            int c = (e & 7) * 8;
            async_copy16(Bm + (size_t)(k0 + r) * Ndim + colBase + c,
                         lds_off(&lB[r * SB + c]));
        }
        async_wait0();
        __syncthreads();

        // A fragments: two 16-row strips of this wave's 32-row band
        v16bf af[2];
        #pragma unroll
        for (int mi = 0; mi < 2; ++mi)
            #pragma unroll
            for (int j = 0; j < 16; ++j)
                af[mi][j] = us_to_bf16(lA[(wave * 32 + mi * 16 + ln) * SA + a_kidx(j, hi)]);

        #pragma unroll
        for (int nc = 0; nc < 4; ++nc) {
            v16bf bfg;
            #pragma unroll
            for (int j = 0; j < 16; ++j)
                bfg[j] = us_to_bf16(lB[(hi * 16 + j) * SB + nc * 16 + ln]);
            #pragma unroll
            for (int mi = 0; mi < 2; ++mi)
                acc[mi][nc] = wmma_bf16(af[mi], bfg, acc[mi][nc]);
        }
    }

    // epilogue: C layout row = v + 8*hi, col = ln (per 16x16 tile)
    #pragma unroll
    for (int mi = 0; mi < 2; ++mi)
        #pragma unroll
        for (int nc = 0; nc < 4; ++nc)
            #pragma unroll
            for (int v = 0; v < 8; ++v) {
                int row = rowBase + wave * 32 + mi * 16 + v + 8 * hi;
                int col = colBase + nc * 16 + ln;
                if (OUT_BF16)
                    ((unsigned short*)Cp)[(size_t)row * Ndim + col] =
                        f32_to_bf16_bits(acc[mi][nc][v]);
                else
                    ((float*)Cp)[(size_t)row * Ndim + col] = acc[mi][nc][v];
            }
}

// ---------------- flash attention (causal) ----------------
// qkv layout: [B,T,3,H,HD] bf16 bits. Each wave owns a 16-row query tile.
// Block = 128 threads (4 waves) -> 64 query rows; grid = (T/64, B*H).
__global__ __launch_bounds__(128)
void flash_attn_kernel(const unsigned short* __restrict__ qkv,
                       unsigned short* __restrict__ ctx) {
    __shared__ unsigned short lP[4][16 * 33];   // per-wave P scratch (16x32)

    const int tid  = threadIdx.x;
    const int lane = tid & 31;
    const int wave = tid >> 5;
    const int hi   = lane >> 4;
    const int ln   = lane & 15;
    const int bh = blockIdx.y;
    const int b  = bh / HH, h = bh % HH;
    const int qbase = blockIdx.x * 64 + wave * 16;

    const size_t rs = NQKV;  // token row stride
    const unsigned short* Qb = qkv + (size_t)b * TT * NQKV + (size_t)h * HDD;
    const unsigned short* Kb = Qb + (size_t)HH * HDD;       // s=1 slice
    const unsigned short* Vb = Qb + (size_t)2 * HH * HDD;   // s=2 slice
    const float scale = 0.088388347648318447f;              // 1/sqrt(128)

    // Q fragments: 4 chunks of K=32 over HD=128
    v16bf qf[4];
    #pragma unroll
    for (int dc = 0; dc < 4; ++dc)
        #pragma unroll
        for (int j = 0; j < 16; ++j)
            qf[dc][j] = us_to_bf16(Qb[(size_t)(qbase + ln) * rs + dc * 32 + a_kidx(j, hi)]);

    v8f o[8];
    #pragma unroll
    for (int i = 0; i < 8; ++i) o[i] = zero8();
    float m[8], l[8];
    #pragma unroll
    for (int v = 0; v < 8; ++v) { m[v] = -1e30f; l[v] = 0.0f; }

    const int kvend = qbase + 16;                // causal bound
    for (int kv0 = 0; kv0 < kvend; kv0 += 32) {
        // S = Q K^T for a 16x32 score tile (two 16x16 WMMA targets)
        v8f sc0 = zero8(), sc1 = zero8();
        #pragma unroll
        for (int dc = 0; dc < 4; ++dc) {
            v16bf kb0, kb1;
            #pragma unroll
            for (int j = 0; j < 16; ++j) {
                int d = dc * 32 + hi * 16 + j;   // B-frag k index = hi*16+j
                kb0[j] = us_to_bf16(Kb[(size_t)(kv0 + ln) * rs + d]);
                kb1[j] = us_to_bf16(Kb[(size_t)(kv0 + 16 + ln) * rs + d]);
            }
            sc0 = wmma_bf16(qf[dc], kb0, sc0);
            sc1 = wmma_bf16(qf[dc], kb1, sc1);
        }

        // online softmax update; rows live in 16-lane halves of the wave
        unsigned short* myP = lP[wave];
        #pragma unroll
        for (int v = 0; v < 8; ++v) {
            int qrow = qbase + v + 8 * hi;
            float s0 = sc0[v] * scale; if (kv0 + ln      > qrow) s0 = -1e30f;
            float s1 = sc1[v] * scale; if (kv0 + 16 + ln > qrow) s1 = -1e30f;
            float mt = fmaxf(s0, s1);
            #pragma unroll
            for (int off = 1; off < 16; off <<= 1) mt = fmaxf(mt, __shfl_xor(mt, off, 32));
            float mnew = fmaxf(m[v], mt);
            float corr = __expf(m[v] - mnew);
            float p0 = __expf(s0 - mnew);
            float p1 = __expf(s1 - mnew);
            float rsum = p0 + p1;
            #pragma unroll
            for (int off = 1; off < 16; off <<= 1) rsum += __shfl_xor(rsum, off, 32);
            l[v] = l[v] * corr + rsum;
            m[v] = mnew;
            #pragma unroll
            for (int d2 = 0; d2 < 8; ++d2) o[d2][v] = o[d2][v] * corr;
            int r = v + 8 * hi;
            myP[r * 33 + ln]      = f32_to_bf16_bits(p0);
            myP[r * 33 + 16 + ln] = f32_to_bf16_bits(p1);
        }
        // same-wave DS ops are in-order; wait + compiler barrier before re-read
        asm volatile("s_wait_dscnt 0" ::: "memory");

        // P as A-fragment (16x32)
        v16bf pf;
        #pragma unroll
        for (int j = 0; j < 16; ++j)
            pf[j] = us_to_bf16(myP[ln * 33 + a_kidx(j, hi)]);

        // O += P(16x32) * V(32x128) : 8 WMMA over 16-wide d chunks
        #pragma unroll
        for (int d2 = 0; d2 < 8; ++d2) {
            v16bf vf;
            #pragma unroll
            for (int j = 0; j < 16; ++j)
                vf[j] = us_to_bf16(Vb[(size_t)(kv0 + hi * 16 + j) * rs + d2 * 16 + ln]);
            o[d2] = wmma_bf16(pf, vf, o[d2]);
        }
    }

    // epilogue: ctx[b,t,h*HD+d] = O / l  (bf16)
    #pragma unroll
    for (int d2 = 0; d2 < 8; ++d2)
        #pragma unroll
        for (int v = 0; v < 8; ++v) {
            int qrow = qbase + v + 8 * hi;
            float val = o[d2][v] / l[v];
            ctx[(size_t)(b * TT + qrow) * (HH * HDD) + h * HDD + d2 * 16 + ln] =
                f32_to_bf16_bits(val);
        }
}

// ---------------- host launcher ----------------
extern "C" void kernel_launch(void* const* d_in, const int* in_sizes, int n_in,
                              void* d_out, int out_size, void* d_ws, size_t ws_size,
                              hipStream_t stream) {
    const float* x    = (const float*)d_in[0];
    // d_in[1] = causal mask (unused; causality handled analytically)
    const float* wqkv = (const float*)d_in[2];
    const float* wo   = (const float*)d_in[3];
    float* out = (float*)d_out;

    char* p = (char*)d_ws;
    unsigned short* xb   = (unsigned short*)p; p += (size_t)MROWS * DIMM * 2;
    unsigned short* wb   = (unsigned short*)p; p += (size_t)DIMM * NQKV * 2;
    unsigned short* wob  = (unsigned short*)p; p += (size_t)DIMM * DIMM * 2;
    unsigned short* qkvb = (unsigned short*)p; p += (size_t)MROWS * NQKV * 2;
    unsigned short* ctxb = (unsigned short*)p; p += (size_t)MROWS * DIMM * 2;

    const int nx  = MROWS * DIMM;
    const int nw  = DIMM * NQKV;
    const int nwo = DIMM * DIMM;
    f32_to_bf16_kernel<<<(nx  + 255) / 256, 256, 0, stream>>>(x,    xb,  nx);
    f32_to_bf16_kernel<<<(nw  + 255) / 256, 256, 0, stream>>>(wqkv, wb,  nw);
    f32_to_bf16_kernel<<<(nwo + 255) / 256, 256, 0, stream>>>(wo,   wob, nwo);

    // qkv = x @ wqkv   (bf16 out)
    gemm_wmma_bf16<true><<<dim3(NQKV / BN, MROWS / BM), 128, 0, stream>>>(
        xb, wb, (void*)qkvb, MROWS, NQKV, DIMM);

    // causal flash attention -> ctx (bf16)
    flash_attn_kernel<<<dim3(TT / 64, BB * HH), 128, 0, stream>>>(qkvb, ctxb);

    // out = ctx @ wo   (f32 out)
    gemm_wmma_bf16<false><<<dim3(DIMM / BN, MROWS / BM), 128, 0, stream>>>(
        ctxb, wob, (void*)out, MROWS, DIMM, DIMM);
}
